// VisionTransformer_53197464928935
// MI455X (gfx1250) — compile-verified, matrix operand order checked by reading, not run on hardware
//
#include <hip/hip_runtime.h>
#include <hip/hip_bf16.h>
#include <math.h>
#include <stdint.h>

// ---------------- types ----------------
typedef __attribute__((ext_vector_type(16))) _Float16 v16h;
typedef __attribute__((ext_vector_type(8)))  _Float16 v8h;
typedef __attribute__((ext_vector_type(8)))  float    v8f;

#define DEPTH 12
#define DIM   768
#define HEADS 12
#define MLPD  3072
#define NTOK  197          // tokens per image
#define BATCH 32
#define ROWS  (BATCH*NTOK) // 6304
#define HD    64
#define NCLS  1000

// ---------------- fragment helpers ----------------
// 16-bit A-matrix 16x32 layout: row = lane&15, K = (e&7) + (e>>3)*16 + (lane>>4)*8
// B fragment is symmetric (lane holds one column, same K striping).
__device__ __forceinline__ v16h combine_frag(v8h lo, v8h hi) {
  v16h r;
#pragma unroll
  for (int i = 0; i < 8; ++i) { r[i] = lo[i]; r[i + 8] = hi[i]; }
  return r;
}

__device__ __forceinline__ v16h load_frag_lds(const _Float16* base, int khalf) {
  v8h lo = *(const v8h*)(base + khalf * 8);
  v8h hi = *(const v8h*)(base + 16 + khalf * 8);
  return combine_frag(lo, hi);
}

__device__ __forceinline__ v8h cvt8(float4 f0, float4 f1) {
  v8h h;
  h[0] = (_Float16)f0.x; h[1] = (_Float16)f0.y; h[2] = (_Float16)f0.z; h[3] = (_Float16)f0.w;
  h[4] = (_Float16)f1.x; h[5] = (_Float16)f1.y; h[6] = (_Float16)f1.z; h[7] = (_Float16)f1.w;
  return h;
}

// async 16B global -> LDS copy (CDNA5, ASYNCcnt)
__device__ __forceinline__ void async_copy_b128(const void* gptr, void* lptr) {
  unsigned lds_off = (unsigned)(uintptr_t)lptr;       // flat LDS addr: low 32 bits = LDS offset
  unsigned long long ga = (unsigned long long)(uintptr_t)gptr;
  asm volatile("global_load_async_to_lds_b128 %0, %1, off"
               :: "v"(lds_off), "v"(ga) : "memory");
}
__device__ __forceinline__ void async_wait0() {
  asm volatile("s_wait_asynccnt 0x0" ::: "memory");
}

// ---------------- weight convert + transpose: W[K,N] f32 -> WT[N,K] f16 ----------------
__global__ __launch_bounds__(256) void wconv_kernel(
    const float* __restrict__ W, _Float16* __restrict__ WT, int K, int Nc)
{
  __shared__ _Float16 tile[32][36];   // [n][k], padded
  const int t  = threadIdx.x;
  const int k0 = blockIdx.x * 32;
  const int n0 = blockIdx.y * 32;

  const int lk  = t >> 3;           // 0..31
  const int lnq = (t & 7) * 4;      // 0..28 step 4
  const int nn  = n0 + lnq;
  float4 f = {0.f, 0.f, 0.f, 0.f};
  if (nn + 3 < Nc)                   // Nc % 4 == 0 for all our shapes
    f = *(const float4*)(W + (size_t)(k0 + lk) * Nc + nn);
  tile[lnq + 0][lk] = (_Float16)f.x;
  tile[lnq + 1][lk] = (_Float16)f.y;
  tile[lnq + 2][lk] = (_Float16)f.z;
  tile[lnq + 3][lk] = (_Float16)f.w;
  __syncthreads();

  const int ln  = t >> 3;
  const int lkq = (t & 7) * 4;
  if (n0 + ln < Nc) {
    _Float16* dst = WT + (size_t)(n0 + ln) * K + k0 + lkq;
    dst[0] = tile[ln][lkq + 0];
    dst[1] = tile[ln][lkq + 1];
    dst[2] = tile[ln][lkq + 2];
    dst[3] = tile[ln][lkq + 3];
  }
}

// ---------------- padded pre-scaled attention mask ----------------
#define NPAD 224
__global__ __launch_bounds__(256) void maskprep_kernel(
    const float* __restrict__ mask, float* __restrict__ maskP)
{
  const int q = blockIdx.x;                  // 0..196
  for (int k = threadIdx.x; k < NPAD; k += 256) {
    float v = 0.0f;
    if (k < NTOK) v = 0.125f * mask[(size_t)q * NTOK + k];   // fold 1/sqrt(64)
    maskP[(size_t)q * NPAD + k] = v;
  }
}

// ---------------- generic WMMA GEMM (128x128 block tile) ----------------
// C[M,Nc] = act( A[M,K](f32) * WT[Nc,K](f16, pre-transposed) + bias ) (+ residual)
// FLAGS (compile-time): 1 = bias, 2 = exact GELU, 4 = residual add
#define TILE_M 128
#define TILE_N 128
#define TILE_K 32
#define LSTR   40   // halves per LDS row (80B: 16B-aligned chunks, bank-spread)

template<int FLAGS>
__global__ __launch_bounds__(256) void gemm_wmma_kernel(
    const float* __restrict__ A, int lda,
    const _Float16* __restrict__ WT,          // row n holds K halves
    const float* __restrict__ bias,
    const float* __restrict__ residual,
    float* __restrict__ C, int ldc,
    int M, int Nc, int K)
{
  __shared__ __align__(16) _Float16 As[TILE_M * LSTR];   // 10 KB
  __shared__ __align__(16) _Float16 Bs[TILE_N * LSTR];   // 10 KB, [n][k]

  const int tid  = threadIdx.x;
  const int lane = tid & 31;
  const int wave = tid >> 5;
  const int m0 = blockIdx.y * TILE_M;
  const int n0 = blockIdx.x * TILE_N;

  const int wm = (wave >> 1) * 32;   // 0,32,64,96
  const int wn = (wave & 1) * 64;    // 0,64

  const int fr    = lane & 15;
  const int khalf = lane >> 4;

  v8f acc[2][4] = {};

  // staging: each thread owns two 8-half chunks of A and of B (rows sr, sr+64)
  const int sr  = tid >> 2;          // 0..63
  const int sc0 = (tid & 3) * 8;     // 0,8,16,24
  int gr0 = m0 + sr;      if (gr0 >= M)  gr0 = M - 1;
  int gr1 = m0 + sr + 64; if (gr1 >= M)  gr1 = M - 1;
  int gn0 = n0 + sr;      if (gn0 >= Nc) gn0 = Nc - 1;
  int gn1 = n0 + sr + 64; if (gn1 >= Nc) gn1 = Nc - 1;
  const float*    aRow0 = A  + (size_t)gr0 * lda + sc0;
  const float*    aRow1 = A  + (size_t)gr1 * lda + sc0;
  const _Float16* bRow0 = WT + (size_t)gn0 * K   + sc0;
  const _Float16* bRow1 = WT + (size_t)gn1 * K   + sc0;
  _Float16* asA0 = &As[sr * LSTR + sc0];
  _Float16* asA1 = &As[(sr + 64) * LSTR + sc0];
  _Float16* lsB0 = &Bs[sr * LSTR + sc0];
  _Float16* lsB1 = &Bs[(sr + 64) * LSTR + sc0];

  for (int k0 = 0; k0 < K; k0 += TILE_K) {
    // B tiles: async global->LDS (pure f16 copy, no VGPR round-trip)
    async_copy_b128(bRow0 + k0, lsB0);
    async_copy_b128(bRow1 + k0, lsB1);

    // A tiles: vectorized fp32 load -> f16 convert -> packed 16B LDS store
    {
      const float4* s0 = (const float4*)(aRow0 + k0);
      const float4* s1 = (const float4*)(aRow1 + k0);
      float4 f00 = s0[0], f01 = s0[1];
      float4 f10 = s1[0], f11 = s1[1];
      *(v8h*)asA0 = cvt8(f00, f01);
      *(v8h*)asA1 = cvt8(f10, f11);
    }
    // speculative prefetch of next K-tile (safe past the end)
    __builtin_prefetch(aRow0 + k0 + TILE_K, 0, 1);
    __builtin_prefetch(aRow1 + k0 + TILE_K, 0, 1);

    async_wait0();
    __syncthreads();

    v16h a0 = load_frag_lds(&As[(wm + fr) * LSTR], khalf);
    v16h a1 = load_frag_lds(&As[(wm + 16 + fr) * LSTR], khalf);
    v16h b0 = load_frag_lds(&Bs[(wn + fr) * LSTR], khalf);
    v16h b1 = load_frag_lds(&Bs[(wn + 16 + fr) * LSTR], khalf);
    v16h b2 = load_frag_lds(&Bs[(wn + 32 + fr) * LSTR], khalf);
    v16h b3 = load_frag_lds(&Bs[(wn + 48 + fr) * LSTR], khalf);

    acc[0][0] = __builtin_amdgcn_wmma_f32_16x16x32_f16(false, a0, false, b0, (short)0, acc[0][0], false, false);
    acc[0][1] = __builtin_amdgcn_wmma_f32_16x16x32_f16(false, a0, false, b1, (short)0, acc[0][1], false, false);
    acc[0][2] = __builtin_amdgcn_wmma_f32_16x16x32_f16(false, a0, false, b2, (short)0, acc[0][2], false, false);
    acc[0][3] = __builtin_amdgcn_wmma_f32_16x16x32_f16(false, a0, false, b3, (short)0, acc[0][3], false, false);
    acc[1][0] = __builtin_amdgcn_wmma_f32_16x16x32_f16(false, a1, false, b0, (short)0, acc[1][0], false, false);
    acc[1][1] = __builtin_amdgcn_wmma_f32_16x16x32_f16(false, a1, false, b1, (short)0, acc[1][1], false, false);
    acc[1][2] = __builtin_amdgcn_wmma_f32_16x16x32_f16(false, a1, false, b2, (short)0, acc[1][2], false, false);
    acc[1][3] = __builtin_amdgcn_wmma_f32_16x16x32_f16(false, a1, false, b3, (short)0, acc[1][3], false, false);
    __syncthreads();
  }

  // ---- epilogue (all FLAGS tests are compile-time) ----
#pragma unroll
  for (int i = 0; i < 2; ++i) {
    const int rbase = m0 + wm + i * 16 + khalf * 8;
#pragma unroll
    for (int j = 0; j < 4; ++j) {
      const int col = n0 + wn + j * 16 + fr;
      if (col >= Nc) continue;
      float bi = 0.0f;
      if (FLAGS & 1) bi = bias[col];
      const float* rptr = (FLAGS & 4) ? residual + (size_t)rbase * ldc + col : nullptr;
      float* cptr = C + (size_t)rbase * ldc + col;
#pragma unroll
      for (int r = 0; r < 8; ++r) {
        if (rbase + r >= M) continue;
        float xv = acc[i][j][r] + bi;
        if (FLAGS & 2) xv = 0.5f * xv * (1.0f + erff(xv * 0.70710678118654752f));
        if (FLAGS & 4) xv += rptr[(size_t)r * ldc];
        cptr[(size_t)r * ldc] = xv;
      }
    }
  }
}

// ---------------- LayerNorm (one block per row of 768) ----------------
__global__ __launch_bounds__(256) void ln_kernel(
    const float* __restrict__ in, float* __restrict__ out,
    const float* __restrict__ g, const float* __restrict__ b)
{
  __shared__ float red[256];
  const int tid = threadIdx.x;
  const float* x = in + (size_t)blockIdx.x * DIM;
  float v0 = x[tid], v1 = x[tid + 256], v2 = x[tid + 512];

  red[tid] = v0 + v1 + v2;
  __syncthreads();
  for (int off = 128; off > 0; off >>= 1) {
    if (tid < off) red[tid] += red[tid + off];
    __syncthreads();
  }
  float mean = red[0] * (1.0f / DIM);
  __syncthreads();

  float d0 = v0 - mean, d1 = v1 - mean, d2 = v2 - mean;
  red[tid] = d0 * d0 + d1 * d1 + d2 * d2;
  __syncthreads();
  for (int off = 128; off > 0; off >>= 1) {
    if (tid < off) red[tid] += red[tid + off];
    __syncthreads();
  }
  float rstd = rsqrtf(red[0] * (1.0f / DIM) + 1e-6f);

  float* y = out + (size_t)blockIdx.x * DIM;
  y[tid]       = d0 * rstd * g[tid]       + b[tid];
  y[tid + 256] = d1 * rstd * g[tid + 256] + b[tid + 256];
  y[tid + 512] = d2 * rstd * g[tid + 512] + b[tid + 512];
}

// ---------------- patchify (im2col) ----------------
__global__ __launch_bounds__(256) void im2col_kernel(
    const float* __restrict__ x, float* __restrict__ out)
{
  const int row = blockIdx.x;         // b*196 + gy*14 + gx
  const int b  = row / 196;
  const int p  = row % 196;
  const int gy = p / 14, gx = p % 14;
  for (int e = threadIdx.x; e < DIM; e += 256) {
    int c  = e >> 8;
    int py = (e >> 4) & 15;
    int px = e & 15;
    out[(size_t)row * DIM + e] =
        x[((size_t)(b * 3 + c) * 224 + (gy * 16 + py)) * 224 + (gx * 16 + px)];
  }
}

// ---------------- assemble tokens: [cls | patches] + pos ----------------
__global__ __launch_bounds__(256) void assemble_kernel(
    const float* __restrict__ pe, const float* __restrict__ cls,
    const float* __restrict__ pos, float* __restrict__ t)
{
  const int row = blockIdx.x;
  const int b = row / NTOK;
  const int n = row % NTOK;
  for (int e = threadIdx.x; e < DIM; e += 256) {
    float v = (n == 0) ? cls[e] : pe[(size_t)(b * 196 + n - 1) * DIM + e];
    t[(size_t)row * DIM + e] = v + pos[(size_t)n * DIM + e];
  }
}

// ---------------- gather cls rows ----------------
__global__ __launch_bounds__(256) void gather_cls_kernel(
    const float* __restrict__ t, float* __restrict__ pooled)
{
  const int b = blockIdx.x;
  for (int e = threadIdx.x; e < DIM; e += 256)
    pooled[(size_t)b * DIM + e] = t[(size_t)(b * NTOK) * DIM + e];
}

// ---------------- WMMA attention: one block (4 waves) per (b, head) ----------------
// LDS-staged K/V (f16), 13 query tiles of 16 processed cooperatively.
#define QT_PER 13
#define KSTR   72           // Kt/Qt row stride (halves)
#define VSTR   232          // Vt/pr row stride (halves)
__global__ __launch_bounds__(128) void attn_kernel(
    const float* __restrict__ qkv, const float* __restrict__ maskP,
    float* __restrict__ o)
{
  __shared__ __align__(16) _Float16 Kt[NPAD * KSTR];   // [key][d]   ~32 KB
  __shared__ __align__(16) _Float16 Vt[HD * VSTR];     // [d][key]   ~29 KB
  __shared__ __align__(16) _Float16 Qt[16 * KSTR];     // [q][d]      2.3 KB
  __shared__ __align__(16) float    sc[16 * NPAD];     // scores     14 KB
  __shared__ __align__(16) _Float16 pr[16 * VSTR];     // probs       7.4 KB
  __shared__ float red[16 * 8];
  __shared__ float rowv[16];

  const int h = blockIdx.x % HEADS;
  const int b = blockIdx.x / HEADS;
  const int tid  = threadIdx.x;
  const int lane = tid & 31;
  const int wave = tid >> 5;
  const int col  = lane & 15;
  const int half = lane >> 4;

  const float* base = qkv + (size_t)(b * NTOK) * (3 * DIM) + h * HD;

  // ---- stage K[key][d] and V^T[d][key] into LDS (f16), keys padded by clamping ----
  {
    const int r0 = tid >> 3;          // 0..15
    const int c0 = (tid & 7) * 8;     // 0..56
    for (int it = 0; it < NPAD / 16; ++it) {
      int key = r0 + it * 16;
      int kcl = key < NTOK ? key : NTOK - 1;
      const float4* ks = (const float4*)(base + (size_t)kcl * (3 * DIM) + DIM + c0);
      float4 k0 = ks[0], k1 = ks[1];
      *(v8h*)&Kt[key * KSTR + c0] = cvt8(k0, k1);
      const float4* vs = (const float4*)(base + (size_t)kcl * (3 * DIM) + 2 * DIM + c0);
      float4 p0 = vs[0], p1 = vs[1];
      v8h vh = cvt8(p0, p1);
#pragma unroll
      for (int i = 0; i < 8; ++i) Vt[(c0 + i) * VSTR + key] = vh[i];
    }
  }
  __syncthreads();

  for (int qt = 0; qt < QT_PER; ++qt) {
    // ---- stage Q tile (16 x 64) ----
    {
      const int q0 = tid >> 3;
      const int c0 = (tid & 7) * 8;
      int qg = qt * 16 + q0;
      int qcl = qg < NTOK ? qg : NTOK - 1;
      const float4* qs = (const float4*)(base + (size_t)qcl * (3 * DIM) + c0);
      float4 f0 = qs[0], f1 = qs[1];
      *(v8h*)&Qt[q0 * KSTR + c0] = cvt8(f0, f1);
    }
    __syncthreads();

    v16h aq0 = load_frag_lds(&Qt[col * KSTR], half);        // d 0..31
    v16h aq1 = load_frag_lds(&Qt[col * KSTR + 32], half);   // d 32..63

    // ---- scores: each wave takes key tiles wave, wave+4, ... ----
    for (int kt = wave; kt < QT_PER; kt += 4) {
      v16h bk0 = load_frag_lds(&Kt[(kt * 16 + col) * KSTR], half);
      v16h bk1 = load_frag_lds(&Kt[(kt * 16 + col) * KSTR + 32], half);
      v8f c = {};
      c = __builtin_amdgcn_wmma_f32_16x16x32_f16(false, aq0, false, bk0, (short)0, c, false, false);
      c = __builtin_amdgcn_wmma_f32_16x16x32_f16(false, aq1, false, bk1, (short)0, c, false, false);
      const int key = kt * 16 + col;           // < 224: always in-bounds in maskP
#pragma unroll
      for (int r = 0; r < 8; ++r) {
        int rr = half * 8 + r;
        int qg = qt * 16 + rr;
        int qm = qg < NTOK ? qg : NTOK - 1;
        float sval = c[r] * maskP[(size_t)qm * NPAD + key];   // unconditional load
        sc[rr * NPAD + key] = (key < NTOK) ? sval : -INFINITY; // pure v_cndmask
      }
    }
    for (int i = tid; i < 16 * 16; i += 128)
      sc[(i >> 4) * NPAD + 208 + (i & 15)] = -INFINITY;
    __syncthreads();

    // ---- softmax: 8 threads per row, 28 columns each ----
    {
      const int row  = tid & 15;
      const int part = tid >> 4;        // 0..7
      const int j0   = part * 28;       // 8*28 = 224
      float mx = -INFINITY;
      for (int j = 0; j < 28; ++j) mx = fmaxf(mx, sc[row * NPAD + j0 + j]);
      red[row * 8 + part] = mx;
      __syncthreads();
      if (part == 0) {
        float m = red[row * 8];
        for (int p = 1; p < 8; ++p) m = fmaxf(m, red[row * 8 + p]);
        rowv[row] = m;
      }
      __syncthreads();
      float m = rowv[row];
      float s = 0.0f;
      for (int j = 0; j < 28; ++j) {
        float e = expf(sc[row * NPAD + j0 + j] - m);
        sc[row * NPAD + j0 + j] = e;
        s += e;
      }
      red[row * 8 + part] = s;
      __syncthreads();
      if (part == 0) {
        float t2 = red[row * 8];
        for (int p = 1; p < 8; ++p) t2 += red[row * 8 + p];
        rowv[row] = 1.0f / t2;
      }
      __syncthreads();
      float inv = rowv[row];
      for (int j = 0; j < 28; ++j)
        pr[row * VSTR + j0 + j] = (_Float16)(sc[row * NPAD + j0 + j] * inv);
    }
    __syncthreads();

    // ---- out = P @ V : each wave produces one 16-dim tile ----
    {
      const int dt = wave;              // 0..3
      v8f accv = {};
      for (int kt2 = 0; kt2 < NPAD / 32; ++kt2) {
        v16h a  = load_frag_lds(&pr[col * VSTR + kt2 * 32], half);
        v16h bv = load_frag_lds(&Vt[(dt * 16 + col) * VSTR + kt2 * 32], half);
        accv = __builtin_amdgcn_wmma_f32_16x16x32_f16(false, a, false, bv, (short)0, accv, false, false);
      }
#pragma unroll
      for (int r = 0; r < 8; ++r) {
        int qg = qt * 16 + half * 8 + r;
        if (qg < NTOK)
          o[(size_t)(b * NTOK + qg) * DIM + h * HD + dt * 16 + col] = accv[r];
      }
    }
    __syncthreads();   // protect Qt/sc/pr before next query tile
  }
}

// ---------------- launch ----------------
#define GT(x, t) (((x) + (t) - 1) / (t))

extern "C" void kernel_launch(void* const* d_in, const int* in_sizes, int n_in,
                              void* d_out, int out_size, void* d_ws, size_t ws_size,
                              hipStream_t stream) {
  const float* x        = (const float*)d_in[0];
  const float* cp_mask  = (const float*)d_in[1];
  const float* patch_w  = (const float*)d_in[2];
  const float* patch_b  = (const float*)d_in[3];
  const float* cls_tok  = (const float*)d_in[4];
  const float* pos_emb  = (const float*)d_in[5];
  const float* ln1_g    = (const float*)d_in[6];
  const float* ln1_b    = (const float*)d_in[7];
  const float* qkv_w    = (const float*)d_in[8];
  const float* qkv_b    = (const float*)d_in[9];
  const float* proj_w   = (const float*)d_in[10];
  const float* proj_b   = (const float*)d_in[11];
  const float* ln2_g    = (const float*)d_in[12];
  const float* ln2_b    = (const float*)d_in[13];
  const float* fc1_w    = (const float*)d_in[14];
  const float* fc1_b    = (const float*)d_in[15];
  const float* fc2_w    = (const float*)d_in[16];
  const float* fc2_b    = (const float*)d_in[17];
  const float* normf_g  = (const float*)d_in[18];
  const float* normf_b  = (const float*)d_in[19];
  const float* head_w   = (const float*)d_in[20];
  const float* head_b   = (const float*)d_in[21];
  float* out = (float*)d_out;

  // ---- workspace layout ----
  float* ws = (float*)d_ws;
  const size_t SZ_T   = (size_t)ROWS * DIM;
  const size_t SZ_QKV = (size_t)ROWS * 3 * DIM;
  const size_t SZ_MLP = (size_t)ROWS * MLPD;
  float* T   = ws;
  float* H   = T + SZ_T;
  float* O   = H + SZ_T;
  float* QKV = O + SZ_T;
  float* MLP = QKV + SZ_QKV;                 // also reused for im2col

  _Float16* wt = (_Float16*)(MLP + SZ_MLP);  // f16 transposed weights
  _Float16* PATCH_WT = wt;
  _Float16* QKV_WT   = PATCH_WT + (size_t)DIM * DIM;
  _Float16* PROJ_WT  = QKV_WT   + (size_t)DEPTH * 3 * DIM * DIM;
  _Float16* FC1_WT   = PROJ_WT  + (size_t)DEPTH * DIM * DIM;
  _Float16* FC2_WT   = FC1_WT   + (size_t)DEPTH * MLPD * DIM;
  _Float16* HEAD_WT  = FC2_WT   + (size_t)DEPTH * DIM * MLPD;
  float* MASKP = (float*)(HEAD_WT + (size_t)NCLS * DIM);   // 197 x 224 fp32

  const dim3 blk256(256), blk128(128);

  // ---- 0) weights fp32 -> f16 transposed, padded mask (once per call) ----
  maskprep_kernel<<<NTOK, blk256, 0, stream>>>(cp_mask, MASKP);
  wconv_kernel<<<dim3(DIM / 32, DIM / 32), blk256, 0, stream>>>(patch_w, PATCH_WT, DIM, DIM);
  for (int i = 0; i < DEPTH; ++i) {
    wconv_kernel<<<dim3(DIM / 32, 3 * DIM / 32), blk256, 0, stream>>>(
        qkv_w + (size_t)i * DIM * 3 * DIM, QKV_WT + (size_t)i * 3 * DIM * DIM, DIM, 3 * DIM);
    wconv_kernel<<<dim3(DIM / 32, DIM / 32), blk256, 0, stream>>>(
        proj_w + (size_t)i * DIM * DIM, PROJ_WT + (size_t)i * DIM * DIM, DIM, DIM);
    wconv_kernel<<<dim3(DIM / 32, MLPD / 32), blk256, 0, stream>>>(
        fc1_w + (size_t)i * DIM * MLPD, FC1_WT + (size_t)i * MLPD * DIM, DIM, MLPD);
    wconv_kernel<<<dim3(MLPD / 32, DIM / 32), blk256, 0, stream>>>(
        fc2_w + (size_t)i * MLPD * DIM, FC2_WT + (size_t)i * DIM * MLPD, MLPD, DIM);
  }
  wconv_kernel<<<dim3(DIM / 32, GT(NCLS, 32)), blk256, 0, stream>>>(head_w, HEAD_WT, DIM, NCLS);

  // ---- 1) patchify into MLP region (6272 x 768) ----
  im2col_kernel<<<BATCH * 196, blk256, 0, stream>>>(x, MLP);

  // ---- 2) patch embedding GEMM -> QKV region (tmp 6272 x 768) ----
  gemm_wmma_kernel<1><<<dim3(GT(DIM, TILE_N), GT(BATCH * 196, TILE_M)), blk256, 0, stream>>>(
      MLP, DIM, PATCH_WT, patch_b, nullptr, QKV, DIM,
      BATCH * 196, DIM, DIM);

  // ---- 3) assemble tokens + positional embedding -> T ----
  assemble_kernel<<<ROWS, blk256, 0, stream>>>(QKV, cls_tok, pos_emb, T);

  const dim3 gQKV(GT(3 * DIM, TILE_N), GT(ROWS, TILE_M));
  const dim3 gD  (GT(DIM, TILE_N),     GT(ROWS, TILE_M));
  const dim3 gMLP(GT(MLPD, TILE_N),    GT(ROWS, TILE_M));

  for (int i = 0; i < DEPTH; ++i) {
    ln_kernel<<<ROWS, blk256, 0, stream>>>(T, H, ln1_g + i * DIM, ln1_b + i * DIM);

    gemm_wmma_kernel<1><<<gQKV, blk256, 0, stream>>>(
        H, DIM, QKV_WT + (size_t)i * 3 * DIM * DIM,
        qkv_b + (size_t)i * 3 * DIM, nullptr, QKV, 3 * DIM,
        ROWS, 3 * DIM, DIM);

    attn_kernel<<<BATCH * HEADS, blk128, 0, stream>>>(QKV, MASKP, O);

    gemm_wmma_kernel<1 | 4><<<gD, blk256, 0, stream>>>(
        O, DIM, PROJ_WT + (size_t)i * DIM * DIM,
        proj_b + (size_t)i * DIM, T, T, DIM,
        ROWS, DIM, DIM);

    ln_kernel<<<ROWS, blk256, 0, stream>>>(T, H, ln2_g + i * DIM, ln2_b + i * DIM);

    gemm_wmma_kernel<1 | 2><<<gMLP, blk256, 0, stream>>>(
        H, DIM, FC1_WT + (size_t)i * MLPD * DIM,
        fc1_b + (size_t)i * MLPD, nullptr, MLP, MLPD,
        ROWS, MLPD, DIM);

    gemm_wmma_kernel<1 | 4><<<gD, blk256, 0, stream>>>(
        MLP, MLPD, FC2_WT + (size_t)i * DIM * MLPD,
        fc2_b + (size_t)i * DIM, T, T, DIM,
        ROWS, DIM, MLPD);
  }

  // ---- 4) pooled = LN(t[:,0]) @ head_w + head_b ----
  gather_cls_kernel<<<BATCH, blk256, 0, stream>>>(T, H);           // H: 32x768
  ln_kernel<<<BATCH, blk256, 0, stream>>>(H, O, normf_g, normf_b); // O: 32x768
  gemm_wmma_kernel<1><<<dim3(GT(NCLS, TILE_N), 1), blk256, 0, stream>>>(
      O, DIM, HEAD_WT, head_b, nullptr, out, NCLS,
      BATCH, NCLS, DIM);
}
